// DeepRNN_77060303225292
// MI455X (gfx1250) — matrix operand order repaired and checked by reading, not compiled, not run on hardware
//
#include <hip/hip_runtime.h>
#include <hip/hip_bf16.h>

// ---------------------------------------------------------------------------
// DeepRNN for MI455X (gfx1250):
//   phase 0: P0 = emb[x] @ W_ih[0]^T + biases   (parallel WMMA GEMM, 4.3 GF)
//   phase 1: sequential scan, 1 WGP, layer0 K=512 / layer1 K=1024, WMMA bf16
//   phase 2: out = Hs @ W_out^T + b_out  (268 GFLOP, async-LDS double buffer,
//            non-temporal 1 GB output stream so W_out stays L2-resident)
// ---------------------------------------------------------------------------

typedef __attribute__((ext_vector_type(16))) __bf16 v16bf;
typedef __attribute__((ext_vector_type(8)))  float  v8f;

constexpr int Bz = 4;
constexpr int Tz = 2048;
constexpr int Vz = 32000;
constexpr int Hz = 512;
constexpr int Lz = 2;
constexpr int K2 = 2 * Hz;      // 1024 (x concat h)
constexpr int Rz = Tz * Bz;     // 8192 rows (t*B + b)

__device__ inline unsigned short f32_to_bf16_bits(float f) {
    union { float f; unsigned int u; } v; v.f = f;
    unsigned int u = v.u;
    unsigned int r = u + 0x7FFFu + ((u >> 16) & 1u);   // RNE
    return (unsigned short)(r >> 16);
}
__device__ inline __bf16 bf16_from_bits(unsigned short b) {
    union { unsigned short u; __bf16 h; } v; v.u = b; return v.h;
}

// A fragment (16x32 bf16): lane holds row m=lane%16; p points at that row's
// 32-element K-chunk. k_local = (v/4)*16 + (lane/16)*8 + (v%4)*2 + e.
__device__ inline v16bf frag_a(const unsigned short* p, int halfk) {
    v16bf a;
    #pragma unroll
    for (int v = 0; v < 8; ++v) {
        int kl = ((v >> 2) << 4) + (halfk << 3) + ((v & 3) << 1);
        unsigned int w = *(const unsigned int*)(p + kl);
        a[2 * v]     = bf16_from_bits((unsigned short)(w & 0xFFFFu));
        a[2 * v + 1] = bf16_from_bits((unsigned short)(w >> 16));
    }
    return a;
}
// B fragment (32x16 bf16): lane holds column n=lane%16; p points at that
// column's row (N-major storage) K-chunk. k_local = (v/4)*16 + (lane/16)*8 + (v%4)*2 + e.
__device__ inline v16bf frag_b(const unsigned short* p, int halfk) {
    v16bf b;
    #pragma unroll
    for (int v = 0; v < 8; ++v) {
        int kl = ((v >> 2) << 4) + (halfk << 3) + ((v & 3) << 1);
        unsigned int w = *(const unsigned int*)(p + kl);
        b[2 * v]     = bf16_from_bits((unsigned short)(w & 0xFFFFu));
        b[2 * v + 1] = bf16_from_bits((unsigned short)(w >> 16));
    }
    return b;
}
__device__ inline v16bf frag_zero() {
    v16bf z;
    #pragma unroll
    for (int v = 0; v < 16; ++v) z[v] = bf16_from_bits(0);
    return z;
}

// gfx1250 async memory->LDS copy (ASYNCcnt tracked). LDS operand = low 32
// bits of the generic pointer (LDS aperture maps addr[31:0] directly).
__device__ inline void async_load_b128(unsigned lds_off, const void* gaddr) {
    asm volatile("global_load_async_to_lds_b128 %0, %1, off"
                 :: "v"(lds_off), "v"((unsigned long long)(size_t)gaddr)
                 : "memory");
}

// ---------------------------------------------------------------------------
// Prep: W_ih[l] ‖ W_hh[l] -> bf16 [L][H(N)][2H(K)] ; W_out -> bf16 [V][H]
// ---------------------------------------------------------------------------
__global__ void prep_wcat_kernel(const float* __restrict__ W_ih,
                                 const float* __restrict__ W_hh,
                                 unsigned short* __restrict__ Wcat) {
    int idx = blockIdx.x * blockDim.x + threadIdx.x;
    if (idx >= Lz * Hz * K2) return;
    int l   = idx / (Hz * K2);
    int rem = idx - l * (Hz * K2);
    int i   = rem / K2;
    int j   = rem - i * K2;
    float v = (j < Hz) ? W_ih[(l * Hz + i) * Hz + j]
                       : W_hh[(l * Hz + i) * Hz + (j - Hz)];
    Wcat[idx] = f32_to_bf16_bits(v);
}

__global__ void prep_wout_kernel(const float* __restrict__ W_out,
                                 unsigned short* __restrict__ Wb) {
    int idx = blockIdx.x * blockDim.x + threadIdx.x;
    if (idx >= Vz * Hz) return;
    Wb[idx] = f32_to_bf16_bits(W_out[idx]);
}

// ---------------------------------------------------------------------------
// Phase 0: P0[r, n] = emb[x[r]] . W_ih[0][n,:] + b_ih[0][n] + b_hh[0][n]
// Block = 512 threads (16 waves, 4M x 4N), block tile 64x256, wave tile 16x64.
// ---------------------------------------------------------------------------
constexpr int MT = 64, NT = 256, KT = 32;

__global__ __launch_bounds__(512) void xproj_kernel(
        const int*   __restrict__ x,       // [B][T]
        const float* __restrict__ embed,   // [V][H]
        const unsigned short* __restrict__ Wcat,  // [L][H][2H] (use l=0, k<H)
        const float* __restrict__ b_ih,
        const float* __restrict__ b_hh,
        float* __restrict__ P0)            // [R][H]
{
    __shared__ __align__(16) unsigned short As[MT * KT];
    __shared__ __align__(16) unsigned short Bs[NT * KT];

    const int tid = threadIdx.x;
    const int wave = tid >> 5, lane = tid & 31;
    const int mw = wave >> 2, nw = wave & 3;
    const int n16 = lane & 15, halfk = lane >> 4;
    const int mBlock = blockIdx.y * MT;
    const int nBlock = blockIdx.x * NT;

    v8f acc[4] = {};
    for (int kc = 0; kc < Hz / KT; ++kc) {
        // A tile: gather embedding rows, convert to bf16 (64x32 = 2048 elems)
        {
            int row = tid >> 3;             // 8 threads per row
            int k0  = (tid & 7) << 2;       // 4 consecutive k each
            int r   = mBlock + row;
            int tok = x[(r & 3) * Tz + (r >> 2)];
            const float* src = embed + (size_t)tok * Hz + kc * KT + k0;
            #pragma unroll
            for (int q = 0; q < 4; ++q)
                As[row * KT + k0 + q] = f32_to_bf16_bits(src[q]);
        }
        // B tile: 256x32 bf16 via uint4 (1024 loads, 2 per thread)
        {
            uint4* Bs128 = (uint4*)Bs;
            int i = tid;
            #pragma unroll
            for (int rr = 0; rr < 2; ++rr, i += 512) {
                int row = i >> 2, q = i & 3;
                Bs128[i] = *(const uint4*)(Wcat + (size_t)(nBlock + row) * K2
                                           + kc * KT + q * 8);
            }
        }
        __syncthreads();
        v16bf a = frag_a(As + (mw * 16 + n16) * KT, halfk);
        #pragma unroll
        for (int j = 0; j < 4; ++j) {
            v16bf b = frag_b(Bs + (nw * 64 + j * 16 + n16) * KT, halfk);
            acc[j] = __builtin_amdgcn_wmma_f32_16x16x32_bf16(
                        false, a, false, b, (short)0, acc[j], false, false);
        }
        __syncthreads();
    }
    #pragma unroll
    for (int j = 0; j < 4; ++j) {
        int n_g = nBlock + nw * 64 + j * 16 + n16;
        float bias = b_ih[n_g] + b_hh[n_g];
        #pragma unroll
        for (int v = 0; v < 8; ++v) {
            int r = mBlock + mw * 16 + v + 8 * halfk;
            P0[(size_t)r * Hz + n_g] = acc[j][v] + bias;
        }
    }
}

// ---------------------------------------------------------------------------
// Phase 1: sequential scan on one WGP (32 waves; wave w owns H-columns
// [16w,16w+16)). Layer0: h0 = tanh(P0[t] + h0 @ W_hh0^T), K=512.
// Layer1: h1 = tanh(h0 @ W_ih1^T + h1 @ W_hh1^T + bias), K=1024.
// Two interleaved accumulators halve the serial WMMA dependency chain.
// ---------------------------------------------------------------------------
__global__ __launch_bounds__(1024) void rnn_scan_kernel(
        const float* __restrict__ P0,      // [R][H] incl. layer-0 biases
        const float* __restrict__ b_ih,
        const float* __restrict__ b_hh,
        const unsigned short* __restrict__ Wcat,  // [L][H][2H]
        unsigned short* __restrict__ Hs,   // [R][H] bf16
        float* __restrict__ h_final)       // [L][B][H]
{
    __shared__ __align__(16) unsigned short hb[Lz][Bz * Hz];  // bf16 hidden
    __shared__ float hf[Lz][Bz * Hz];                         // fp32 hidden

    const int tid   = threadIdx.x;
    const int wave  = tid >> 5;
    const int lane  = tid & 31;
    const int m16   = lane & 15;
    const int halfk = lane >> 4;
    const int ncol  = wave * 16 + m16;

    for (int i = tid; i < Bz * Hz; i += 1024) {
        hb[0][i] = 0; hb[1][i] = 0;
        hf[0][i] = 0.0f; hf[1][i] = 0.0f;
    }
    __syncthreads();

    const unsigned short* w0 = Wcat + (size_t)ncol * K2 + Hz;        // W_hh[0]
    const unsigned short* w1 = Wcat + (size_t)(Hz + ncol) * K2;      // layer 1
    const float bias1 = b_ih[Hz + ncol] + b_hh[Hz + ncol];

    for (int t = 0; t < Tz; ++t) {
        // ---------------- layer 0 (K = 512, 16 chunks, 2 accumulators) -----
        {
            v8f acc0 = {}, acc1 = {};
            for (int c = 0; c < 16; c += 2) {
                v16bf a0 = (m16 < Bz) ? frag_a(hb[0] + m16 * Hz + c * 32, halfk)
                                      : frag_zero();
                v16bf b0 = frag_b(w0 + c * 32, halfk);
                acc0 = __builtin_amdgcn_wmma_f32_16x16x32_bf16(
                         false, a0, false, b0, (short)0, acc0, false, false);
                v16bf a1 = (m16 < Bz) ? frag_a(hb[0] + m16 * Hz + (c + 1) * 32, halfk)
                                      : frag_zero();
                v16bf b1 = frag_b(w0 + (c + 1) * 32, halfk);
                acc1 = __builtin_amdgcn_wmma_f32_16x16x32_bf16(
                         false, a1, false, b1, (short)0, acc1, false, false);
            }
            v8f acc = acc0 + acc1;
            __syncthreads();                 // all reads of old hb[0] done
            if (halfk == 0) {
                #pragma unroll
                for (int v = 0; v < Bz; ++v) {
                    float val = tanhf(acc[v] + P0[((size_t)t * Bz + v) * Hz + ncol]);
                    hf[0][v * Hz + ncol] = val;
                    hb[0][v * Hz + ncol] = f32_to_bf16_bits(val);
                }
            }
            __syncthreads();                 // new h0 visible
        }
        // ---------------- layer 1 (K = 1024, 32 chunks, 2 accumulators) ----
        {
            v8f acc0 = {}, acc1 = {};
            for (int c = 0; c < 32; c += 2) {
                const unsigned short* s0 = (c < 16)
                    ? hb[0] + m16 * Hz + c * 32
                    : hb[1] + m16 * Hz + (c - 16) * 32;
                v16bf a0 = (m16 < Bz) ? frag_a(s0, halfk) : frag_zero();
                v16bf b0 = frag_b(w1 + c * 32, halfk);
                acc0 = __builtin_amdgcn_wmma_f32_16x16x32_bf16(
                         false, a0, false, b0, (short)0, acc0, false, false);
                const unsigned short* s1 = (c + 1 < 16)
                    ? hb[0] + m16 * Hz + (c + 1) * 32
                    : hb[1] + m16 * Hz + (c + 1 - 16) * 32;
                v16bf a1 = (m16 < Bz) ? frag_a(s1, halfk) : frag_zero();
                v16bf b1 = frag_b(w1 + (c + 1) * 32, halfk);
                acc1 = __builtin_amdgcn_wmma_f32_16x16x32_bf16(
                         false, a1, false, b1, (short)0, acc1, false, false);
            }
            v8f acc = acc0 + acc1;
            __syncthreads();                 // all reads of old hb[1] done
            if (halfk == 0) {
                #pragma unroll
                for (int v = 0; v < Bz; ++v) {
                    float val = tanhf(acc[v] + bias1);
                    hf[1][v * Hz + ncol] = val;
                    unsigned short bits = f32_to_bf16_bits(val);
                    hb[1][v * Hz + ncol] = bits;
                    Hs[((size_t)t * Bz + v) * Hz + ncol] = bits;
                }
            }
            __syncthreads();                 // new h1 visible
        }
    }

    for (int i = tid; i < Lz * Bz * Hz; i += 1024) {
        int l = i / (Bz * Hz);
        h_final[i] = hf[l][i - l * (Bz * Hz)];
    }
}

// ---------------------------------------------------------------------------
// Phase 2: out[r, n] = Hs[r,:] . Wout[n,:] + b_out[n], r = t*B + b.
// Block tile 64x256, wave tile 16x64. Tiles staged with async global->LDS
// loads (double buffered, ASYNCcnt); output stored non-temporally so the
// 1 GB stream does not evict the L2-resident W_out/Hs.
// ---------------------------------------------------------------------------
__global__ __launch_bounds__(512) void gemm_out_kernel(
        const unsigned short* __restrict__ Hs,    // [R][H] bf16
        const unsigned short* __restrict__ Wout,  // [V][H] bf16
        const float* __restrict__ b_out,
        float* __restrict__ out)                  // [B][T][V]
{
    __shared__ __align__(16) unsigned short As[2][MT * KT];   // 2 x 4 KB
    __shared__ __align__(16) unsigned short Bs[2][NT * KT];   // 2 x 16 KB

    const int tid = threadIdx.x;
    const int wave = tid >> 5, lane = tid & 31;
    const int mw = wave >> 2, nw = wave & 3;
    const int n16 = lane & 15, halfk = lane >> 4;
    const int mBlock = blockIdx.y * MT;
    const int nBlock = blockIdx.x * NT;
    const int row = tid >> 2, q = tid & 3;   // tile-load coordinates

    // issue one K-chunk's tiles (A: 256 x b128 by waves 0-7; B: 1024 x b128)
    auto issue = [&](int kc, int buf) {
        unsigned ldsA = (unsigned)(size_t)&As[buf][0] + tid * 16;
        unsigned ldsB = (unsigned)(size_t)&Bs[buf][0] + tid * 16;
        if (tid < 256)
            async_load_b128(ldsA, Hs + (size_t)(mBlock + row) * Hz + kc * KT + q * 8);
        async_load_b128(ldsB,
                        Wout + (size_t)(nBlock + row) * Hz + kc * KT + q * 8);
        async_load_b128(ldsB + 512 * 16,
                        Wout + (size_t)(nBlock + row + 128) * Hz + kc * KT + q * 8);
    };

    v8f acc[4] = {};
    issue(0, 0);
    for (int kc = 0; kc < Hz / KT; ++kc) {
        const int cur = kc & 1;
        if (kc + 1 < Hz / KT) {
            issue(kc + 1, cur ^ 1);
            // async loads complete in order: waiting down to the newly issued
            // batch's op count guarantees batch kc has landed in LDS.
            if (tid < 256) asm volatile("s_wait_asynccnt 0x3" ::: "memory");
            else           asm volatile("s_wait_asynccnt 0x2" ::: "memory");
        } else {
            asm volatile("s_wait_asynccnt 0x0" ::: "memory");
        }
        __syncthreads();                       // LDS tiles visible to all waves
        v16bf a = frag_a(As[cur] + (mw * 16 + n16) * KT, halfk);
        #pragma unroll
        for (int j = 0; j < 4; ++j) {
            v16bf b = frag_b(Bs[cur] + (nw * 64 + j * 16 + n16) * KT, halfk);
            acc[j] = __builtin_amdgcn_wmma_f32_16x16x32_bf16(
                        false, a, false, b, (short)0, acc[j], false, false);
        }
        __syncthreads();                       // done reading before overwrite
    }
    #pragma unroll
    for (int j = 0; j < 4; ++j) {
        int n_g = nBlock + nw * 64 + j * 16 + n16;
        float bo = b_out[n_g];
        #pragma unroll
        for (int v = 0; v < 8; ++v) {
            int r = mBlock + mw * 16 + v + 8 * halfk;  // r = t*B + b
            __builtin_nontemporal_store(
                acc[j][v] + bo,
                &out[((size_t)(r & 3) * Tz + (r >> 2)) * Vz + n_g]);
        }
    }
}

// ---------------------------------------------------------------------------
extern "C" void kernel_launch(void* const* d_in, const int* in_sizes, int n_in,
                              void* d_out, int out_size, void* d_ws, size_t ws_size,
                              hipStream_t stream) {
    (void)in_sizes; (void)n_in; (void)out_size; (void)ws_size;
    const int*   x     = (const int*)  d_in[0];
    const float* embed = (const float*)d_in[1];
    const float* W_ih  = (const float*)d_in[2];
    const float* b_ih  = (const float*)d_in[3];
    const float* W_hh  = (const float*)d_in[4];
    const float* b_hh  = (const float*)d_in[5];
    const float* W_out = (const float*)d_in[6];
    const float* b_out = (const float*)d_in[7];
    float* out = (float*)d_out;

    // workspace: Wcat bf16 (2 MB) | Wout bf16 (32.8 MB) | Hs bf16 (8.4 MB)
    //            | P0 fp32 (16.8 MB)
    unsigned short* Wcat  = (unsigned short*)d_ws;
    unsigned short* Woutb = Wcat  + (size_t)Lz * Hz * K2;
    unsigned short* Hs    = Woutb + (size_t)Vz * Hz;
    float*          P0    = (float*)(Hs + (size_t)Rz * Hz);
    float* h_final = out + (size_t)Bz * Tz * Vz;   // second tuple element

    prep_wcat_kernel<<<(Lz * Hz * K2 + 255) / 256, 256, 0, stream>>>(W_ih, W_hh, Wcat);
    prep_wout_kernel<<<(Vz * Hz + 255) / 256, 256, 0, stream>>>(W_out, Woutb);

    dim3 gridP(Hz / NT, Rz / MT);          // 2 x 128
    xproj_kernel<<<gridP, 512, 0, stream>>>(x, embed, Wcat, b_ih, b_hh, P0);

    rnn_scan_kernel<<<1, 1024, 0, stream>>>(P0, b_ih, b_hh, Wcat, Hs, h_final);

    dim3 gridO(Vz / NT, Rz / MT);          // 125 x 128
    gemm_out_kernel<<<gridO, 512, 0, stream>>>(Hs, Woutb, b_out, out);
}
